// GeometricTransformerBlock_53145925321273
// MI455X (gfx1250) — compile-verified, matrix operand order checked by reading, not run on hardware
//
#include <hip/hip_runtime.h>
#include <hip/hip_bf16.h>

// ---------------------------------------------------------------------------
// GeometricTransformerBlock on MI455X (gfx1250, wave32, WMMA bf16, async-LDS)
// ---------------------------------------------------------------------------
#define B_   2
#define N_   2048
#define D_   512
#define H_   8
#define HD_  64
#define DFF_ 2048
#define M_   (B_ * N_)       // 4096 total rows
#define QKLD 1024            // row stride of the packed q|k bf16 buffer
#define SCALE_ 0.125f        // 1/sqrt(64)

typedef __attribute__((ext_vector_type(16))) __bf16 v16bf;
typedef __attribute__((ext_vector_type(8)))  __bf16 v8bf;
typedef __attribute__((ext_vector_type(8)))  float  v8f;

static __device__ __forceinline__ v16bf load_frag(const __bf16* p) {
    // Per-lane 16 bf16 elements of a 16x32 WMMA fragment:
    // 8 contiguous at p, 8 contiguous at p+16 (caller applies hl*8 skew).
    v8bf lo = *(const v8bf*)(p);
    v8bf hi = *(const v8bf*)(p + 16);
    return __builtin_shufflevector(lo, hi, 0,1,2,3,4,5,6,7,8,9,10,11,12,13,14,15);
}

static __device__ __forceinline__ v8f wmma_bf16(v16bf a, v16bf b, v8f c) {
    return __builtin_amdgcn_wmma_f32_16x16x32_bf16(
        /*neg_a=*/false, a, /*neg_b=*/false, b,
        /*c_mod=*/(short)0, c, /*reuse_a=*/false, /*reuse_b=*/false);
}

// Async global->LDS copy (VGLOBAL async op, tracked by ASYNCcnt).
static __device__ __forceinline__ void async_load_b128(unsigned lds_off, const void* gaddr) {
    asm volatile("global_load_async_to_lds_b128 %0, %1, off"
                 :: "v"(lds_off), "v"(gaddr) : "memory");
}
static __device__ __forceinline__ void wait_async0() {
    asm volatile("s_wait_asynccnt 0x0" ::: "memory");
}
static __device__ __forceinline__ void wait_async1() {
    asm volatile("s_wait_asynccnt 0x1" ::: "memory");
}

// ---------------------------------------------------------------------------
// Weight cast + transpose: in fp32 [K, Nc] row-major -> out bf16 [Nc, K]
// ---------------------------------------------------------------------------
__global__ void tcast_kernel(const float* __restrict__ in, __bf16* __restrict__ out,
                             int K, int Nc) {
    int idx = blockIdx.x * blockDim.x + threadIdx.x;
    if (idx < K * Nc) {
        int k = idx / Nc;
        int n = idx - k * Nc;
        out[(size_t)n * K + k] = (__bf16)in[idx];
    }
}

// ---------------------------------------------------------------------------
// LayerNorm over 512-wide rows, one wave32 per row, output bf16
// ---------------------------------------------------------------------------
__global__ __launch_bounds__(32) void ln_kernel(const float* __restrict__ x,
                                                const float* __restrict__ g,
                                                const float* __restrict__ bta,
                                                __bf16* __restrict__ out) {
    const int row  = blockIdx.x;
    const int lane = threadIdx.x;
    const float* xr = x + (size_t)row * D_;

    float vals[16];
    float s = 0.f, s2 = 0.f;
#pragma unroll
    for (int t = 0; t < 16; ++t) {
        float v = xr[t * 32 + lane];
        vals[t] = v;
        s  += v;
        s2 += v * v;
    }
#pragma unroll
    for (int off = 16; off >= 1; off >>= 1) {
        s  += __shfl_xor(s,  off);
        s2 += __shfl_xor(s2, off);
    }
    const float mu  = s  * (1.0f / D_);
    const float var = s2 * (1.0f / D_) - mu * mu;
    const float rs  = rsqrtf(var + 1e-5f);
    __bf16* orow = out + (size_t)row * D_;
#pragma unroll
    for (int t = 0; t < 16; ++t) {
        int c = t * 32 + lane;
        orow[c] = (__bf16)((vals[t] - mu) * rs * g[c] + bta[c]);
    }
}

// ---------------------------------------------------------------------------
// WMMA GEMM:  C[M, Nc] = A[M, K](bf16) * Bt[Nc, K]^T(bf16) + bias
//
// Block = 8 waves, block tile = 256(M) x 64(N); each wave owns 32(M) x 64(N).
// The 64x32 B tile (shared by all 8 waves) is staged global->LDS with
// double-buffered global_load_async_to_lds_b128 (ASYNCcnt), then read back as
// WMMA B-fragments with ds_load_b128. A-fragments stream from global (L1/L2).
//
// Epilogues:
//   0 = QKV  : n<1024 -> bf16 into q|k buffer (ld=1024); n>=1024 -> vT[b,h,d,n]
//   1 = GELU : bf16 exact-gelu into outH (ld = Nc)
//   2 = RESID: fp32 (val + resid) into outF (ld = Nc)
// ---------------------------------------------------------------------------
#define EPI_QKV   0
#define EPI_GELU  1
#define EPI_RESID 2

template <int EPI>
__global__ __launch_bounds__(256) void gemm_kernel(
    const __bf16* __restrict__ A, const __bf16* __restrict__ Bt,
    const float* __restrict__ bias, const float* __restrict__ resid,
    float* __restrict__ outF, __bf16* __restrict__ outH,
    __bf16* __restrict__ vT, int K, int Nc) {

    __shared__ __align__(16) __bf16 sB[2][64 * 32];   // two 4 KB B-tile buffers

    const int lane = threadIdx.x & 31;
    const int wave = threadIdx.x >> 5;
    const int hl = lane >> 4;
    const int ln = lane & 15;

    const int m0 = blockIdx.y * 256 + wave * 32;   // wave's 32-row slab
    const int n0 = blockIdx.x * 64;                // block's 64-col slab

    // B-tile staging: thread t copies 16 B of row (t>>2), chunk (t&3).
    const int srow = threadIdx.x >> 2;
    const int schk = threadIdx.x & 3;
    const __bf16* gsrc = Bt + (size_t)(n0 + srow) * K + schk * 8;
    const unsigned lbase = (unsigned)(unsigned long long)(&sB[0][0]);
    const unsigned loff  = (unsigned)(srow * 64 + schk * 16);   // bytes in tile

    v8f acc[2][4] = {};
    const __bf16* arow0 = A + (size_t)(m0 + ln) * K + hl * 8;
    const __bf16* arow1 = arow0 + (size_t)16 * K;

    const int T = K >> 5;                          // K/32 k-steps
    async_load_b128(lbase + loff, gsrc);           // prefetch tile 0 -> buf 0

    for (int i = 0; i < T; ++i) {
        if (i + 1 < T) {                           // prefetch next tile
            async_load_b128(lbase + ((i + 1) & 1) * 4096u + loff,
                            gsrc + (size_t)(i + 1) * 32);
            wait_async1();                         // tile i landed (in-order)
        } else {
            wait_async0();
        }
        __syncthreads();

        const __bf16* bbuf = &sB[i & 1][0];
        v16bf a0 = load_frag(arow0 + i * 32);
        v16bf a1 = load_frag(arow1 + i * 32);
#pragma unroll
        for (int j = 0; j < 4; ++j) {
            v16bf bfr = load_frag(bbuf + (j * 16 + ln) * 32 + hl * 8);
            acc[0][j] = wmma_bf16(a0, bfr, acc[0][j]);
            acc[1][j] = wmma_bf16(a1, bfr, acc[1][j]);
        }
        __syncthreads();                           // done reading buf[i&1]
    }

#pragma unroll
    for (int mi = 0; mi < 2; ++mi)
#pragma unroll
        for (int j = 0; j < 4; ++j) {
            const int n = n0 + j * 16 + ln;
            const float bn = bias[n];
#pragma unroll
            for (int r = 0; r < 8; ++r) {
                const int m = m0 + mi * 16 + r + 8 * hl;
                float v = acc[mi][j][r] + bn;
                if constexpr (EPI == EPI_QKV) {
                    if (n < 1024) {
                        outH[(size_t)m * QKLD + n] = (__bf16)v;   // q | k packed
                    } else {
                        const int d  = (n - 1024) & (HD_ - 1);
                        const int hh = (n - 1024) >> 6;
                        const int bb = m >> 11;                   // N_ = 2048
                        const int sq = m & (N_ - 1);
                        vT[(((size_t)bb * H_ + hh) * HD_ + d) * N_ + sq] = (__bf16)v;
                    }
                } else if constexpr (EPI == EPI_GELU) {
                    float gl = 0.5f * v * (1.0f + erff(v * 0.70710678118654752f));
                    outH[(size_t)m * Nc + n] = (__bf16)gl;
                } else {  // EPI_RESID
                    outF[(size_t)m * Nc + n] = v + resid[(size_t)m * Nc + n];
                }
            }
        }
}

// ---------------------------------------------------------------------------
// Geometric-masked flash attention. One wave32 per (b, h, 16-query tile).
// Computes S^T = K_tile x Q^T so that (a) softmax-over-keys reduces across
// registers + one shfl_xor(16), and (b) exp(S^T) lands directly in the WMMA
// A-fragment layout for the P x V product (no transpose / LDS round trip).
// ---------------------------------------------------------------------------
__global__ __launch_bounds__(32) void attn_kernel(
    const __bf16* __restrict__ qk,    // [M, 1024] bf16: q | k packed per row
    const __bf16* __restrict__ vT,    // [B,H,64,N] bf16
    const int* __restrict__ cpos,     // [B,N,2]
    const int* __restrict__ clvl,     // [B,N]
    __bf16* __restrict__ out) {       // [M, 512] bf16

    const int lane = threadIdx.x & 31;
    const int hl = lane >> 4;
    const int ln = lane & 15;
    const int qt = blockIdx.x;        // 0..127
    const int h  = blockIdx.y;        // 0..7
    const int b  = blockIdx.z;        // 0..1

    const __bf16* qbase = qk + (size_t)(b * N_ + qt * 16) * QKLD + h * HD_;
    const __bf16* kbase = qk + (size_t)(b * N_) * QKLD + D_ + h * HD_;
    const __bf16* vbase = vT + ((size_t)(b * H_ + h) * HD_) * (size_t)N_;

    // Q fragments (serve as B-fragments of Q^T); lane = query column.
    v16bf qf[2];
#pragma unroll
    for (int c = 0; c < 2; ++c)
        qf[c] = load_frag(qbase + (size_t)ln * QKLD + c * 32 + hl * 8);

    // This lane's query coords (query index = ln; duplicated in both halves).
    const int qi  = qt * 16 + ln;
    const int qgi = b * N_ + qi;
    const int qpx = cpos[2 * qgi], qpy = cpos[2 * qgi + 1], ql = clvl[qgi];
    const int win = (ql == 0) ? 512 : (ql == 1) ? 64 : 16;
    const bool qok = (ql <= 2);

    float m_run = -1e30f, l_run = 0.f;
    v8f o[4] = {};

    for (int jt = 0; jt < N_ / 32; ++jt) {
        // Cooperative key-coords load: lane -> key (jt*32 + lane).
        const int kgi = b * N_ + jt * 32 + lane;
        const int kpx_l = cpos[2 * kgi];
        const int kpy_l = cpos[2 * kgi + 1];
        const int kl_l  = clvl[kgi];

        // S^T tiles: rows = keys, cols = queries.
        v8f st[2];
#pragma unroll
        for (int s = 0; s < 2; ++s) {
            const __bf16* krow = kbase + (size_t)(jt * 32 + s * 16 + ln) * QKLD + hl * 8;
            v8f a = {};
            a = wmma_bf16(load_frag(krow),      qf[0], a);
            a = wmma_bf16(load_frag(krow + 32), qf[1], a);
            st[s] = a;
        }

        // Mask + online softmax. Element (s, r): key jt*32 + s*16 + r + 8*hl.
        float p[16];
        float tmax = m_run;
#pragma unroll
        for (int s = 0; s < 2; ++s)
#pragma unroll
            for (int r = 0; r < 8; ++r) {
                const int kk = s * 16 + r + 8 * hl;
                const int jg = jt * 32 + kk;
                const int kpx = __shfl(kpx_l, kk);
                const int kpy = __shfl(kpy_l, kk);
                const int kl  = __shfl(kl_l,  kk);
                const int manh = __builtin_abs(qpx - kpx) + __builtin_abs(qpy - kpy);
                const bool ok = ((ql == kl) && (manh <= 1) &&
                                 (__builtin_abs(qi - jg) < win) && qok && (kl <= 2))
                                || (qi == jg);
                const float sv = ok ? st[s][r] * SCALE_ : -1.0e9f;
                p[s * 8 + r] = sv;
                tmax = fmaxf(tmax, sv);
            }
        tmax = fmaxf(tmax, __shfl_xor(tmax, 16));

        const float alpha = __expf(m_run - tmax);
        float lsum = 0.f;
#pragma unroll
        for (int e = 0; e < 16; ++e) {
            const float pe = __expf(p[e] - tmax);
            p[e] = pe;
            lsum += pe;
        }
        lsum += __shfl_xor(lsum, 16);
        l_run = l_run * alpha + lsum;
        m_run = tmax;

        // Rescale O rows (row of O = r + 8*hl -> alpha lives at lane r+8*hl).
#pragma unroll
        for (int r = 0; r < 8; ++r) {
            const float ar = __shfl(alpha, r + 8 * hl);
#pragma unroll
            for (int f = 0; f < 4; ++f) o[f][r] *= ar;
        }

        // P is already in A-fragment layout: lane<16 holds keys {0..7,16..23},
        // lane>=16 holds {8..15,24..31}.
        v16bf pa;
#pragma unroll
        for (int e = 0; e < 8; ++e) {
            pa[e]     = (__bf16)p[e];       // keys s=0 chunk
            pa[8 + e] = (__bf16)p[8 + e];   // keys s=1 chunk
        }
#pragma unroll
        for (int f = 0; f < 4; ++f) {
            v16bf vf = load_frag(vbase + (size_t)(f * 16 + ln) * N_ + jt * 32 + hl * 8);
            o[f] = wmma_bf16(pa, vf, o[f]);
        }
    }

    // Normalize + store bf16 attention output.
#pragma unroll
    for (int r = 0; r < 8; ++r) {
        const float lr  = __shfl(l_run, r + 8 * hl);
        const float inv = 1.0f / lr;
        const int row = qt * 16 + r + 8 * hl;
        __bf16* orow = out + (size_t)(b * N_ + row) * D_ + h * HD_;
#pragma unroll
        for (int f = 0; f < 4; ++f)
            orow[f * 16 + ln] = (__bf16)(o[f][r] * inv);
    }
}

// ---------------------------------------------------------------------------
// Host-side orchestration
// ---------------------------------------------------------------------------
extern "C" void kernel_launch(void* const* d_in, const int* in_sizes, int n_in,
                              void* d_out, int out_size, void* d_ws, size_t ws_size,
                              hipStream_t stream) {
    (void)in_sizes; (void)n_in; (void)out_size; (void)ws_size;

    const float* x     = (const float*)d_in[0];
    const int*   cpos  = (const int*)  d_in[1];
    const int*   clvl  = (const int*)  d_in[2];
    const float* w_qkv = (const float*)d_in[3];
    const float* b_qkv = (const float*)d_in[4];
    const float* w_out = (const float*)d_in[5];
    const float* b_out = (const float*)d_in[6];
    const float* w_ff1 = (const float*)d_in[7];
    const float* b_ff1 = (const float*)d_in[8];
    const float* w_ff2 = (const float*)d_in[9];
    const float* b_ff2 = (const float*)d_in[10];
    const float* ln1g  = (const float*)d_in[11];
    const float* ln1b  = (const float*)d_in[12];
    const float* ln2g  = (const float*)d_in[13];
    const float* ln2b  = (const float*)d_in[14];
    float* outp = (float*)d_out;

    // Workspace carve-up (~54 MiB total).
    char* w = (char*)d_ws;
    auto take = [&](size_t bytes) {
        char* p = w;
        w += (bytes + 255) & ~(size_t)255;
        return p;
    };
    __bf16* xn1    = (__bf16*)take((size_t)M_ * D_   * 2);  // LN1 output
    __bf16* xn2    = (__bf16*)take((size_t)M_ * D_   * 2);  // LN2 output
    __bf16* attn   = (__bf16*)take((size_t)M_ * D_   * 2);  // attention output
    __bf16* qkbuf  = (__bf16*)take((size_t)M_ * QKLD * 2);  // q|k packed
    __bf16* vTbuf  = (__bf16*)take((size_t)B_ * H_ * HD_ * N_ * 2);
    float*  x1     = (float*) take((size_t)M_ * D_   * 4);  // post-attn residual
    __bf16* hbuf   = (__bf16*)take((size_t)M_ * DFF_ * 2);  // GELU hidden
    __bf16* wqkvT  = (__bf16*)take((size_t)D_   * 3 * D_ * 2);
    __bf16* woutT  = (__bf16*)take((size_t)D_   * D_     * 2);
    __bf16* wff1T  = (__bf16*)take((size_t)DFF_ * D_     * 2);
    __bf16* wff2T  = (__bf16*)take((size_t)D_   * DFF_   * 2);

    // 1) bf16 weight transposes
    tcast_kernel<<<(D_ * 3 * D_ + 255) / 256, 256, 0, stream>>>(w_qkv, wqkvT, D_, 3 * D_);
    tcast_kernel<<<(D_ * D_     + 255) / 256, 256, 0, stream>>>(w_out, woutT, D_, D_);
    tcast_kernel<<<(D_ * DFF_   + 255) / 256, 256, 0, stream>>>(w_ff1, wff1T, D_, DFF_);
    tcast_kernel<<<(DFF_ * D_   + 255) / 256, 256, 0, stream>>>(w_ff2, wff2T, DFF_, D_);

    // 2) LN1
    ln_kernel<<<M_, 32, 0, stream>>>(x, ln1g, ln1b, xn1);

    // 3) QKV projection (writes q|k packed + V transposed)
    {
        dim3 g((3 * D_) / 64, M_ / 256);
        gemm_kernel<EPI_QKV><<<g, 256, 0, stream>>>(
            xn1, wqkvT, b_qkv, nullptr, nullptr, qkbuf, vTbuf, D_, 3 * D_);
    }

    // 4) Geometric-masked flash attention
    {
        dim3 g(N_ / 16, H_, B_);
        attn_kernel<<<g, 32, 0, stream>>>(qkbuf, vTbuf, cpos, clvl, attn);
    }

    // 5) Output projection + residual (fp32)
    {
        dim3 g(D_ / 64, M_ / 256);
        gemm_kernel<EPI_RESID><<<g, 256, 0, stream>>>(
            attn, woutT, b_out, x, x1, nullptr, nullptr, D_, D_);
    }

    // 6) LN2
    ln_kernel<<<M_, 32, 0, stream>>>(x1, ln2g, ln2b, xn2);

    // 7) FF1 + exact GELU (bf16)
    {
        dim3 g(DFF_ / 64, M_ / 256);
        gemm_kernel<EPI_GELU><<<g, 256, 0, stream>>>(
            xn2, wff1T, b_ff1, nullptr, nullptr, hbuf, nullptr, D_, DFF_);
    }

    // 8) FF2 + residual -> final fp32 output
    {
        dim3 g(D_ / 64, M_ / 256);
        gemm_kernel<EPI_RESID><<<g, 256, 0, stream>>>(
            hbuf, wff2T, b_ff2, x1, outp, nullptr, nullptr, DFF_, D_);
    }
}
